// MSD_Net_438086664196
// MI455X (gfx1250) — compile-verified, matrix operand order checked by reading, not run on hardware
//
#include <hip/hip_runtime.h>

// ---------------------------------------------------------------------------
// MSD_Net for MI455X (gfx1250): bf16 WMMA GEMMs + fused conv/reduction kernels
// B=32, V=197, T=512, D=768
// Wave tile 32x64 (2x4 WMMA frags) -> ~21 FLOP/B from L2, WMMA-pipe bound.
// ---------------------------------------------------------------------------

#define BB 32
#define VV 197
#define TT 512
#define DD 768

typedef __attribute__((ext_vector_type(16))) __bf16 v16bf;
typedef __attribute__((ext_vector_type(8)))  float  v8f;

// ---- helpers --------------------------------------------------------------

__device__ __forceinline__ __bf16 f2bf(float f) {
  unsigned u = __float_as_uint(f);
  u += 0x7fffu + ((u >> 16) & 1u);          // round-to-nearest-even
  unsigned short h = (unsigned short)(u >> 16);
  return *(__bf16*)&h;
}

__device__ __forceinline__ float wred32(float x) {
  #pragma unroll
  for (int o = 16; o > 0; o >>= 1) x += __shfl_down(x, o, 32);
  return x;
}

__device__ __forceinline__ v8f wmma_bf16(v16bf a, v16bf b, v8f c) {
  return __builtin_amdgcn_wmma_f32_16x16x32_bf16(false, a, false, b,
                                                 (short)0, c, false, false);
}

// 16-bit A/B fragment load (16x32 A layout; B of a row-major NxK "B^T" array
// loads identically): lane<16 -> K {0..7,16..23}, lane>=16 -> K {8..15,24..31}
__device__ __forceinline__ v16bf load_frag(const __bf16* base, int row_base,
                                           int rows, int ld, int kBase, int lane) {
  int m = lane & 15;
  int half = lane >> 4;
  int r = row_base + m;
  r = (r < rows) ? r : rows - 1;            // clamp (stores are masked)
  const __bf16* p = base + (long)r * ld + kBase + half * 8;
  union { v16bf v; uint4 q[2]; } f;
  f.q[0] = *(const uint4*)p;                // global_load_b128
  f.q[1] = *(const uint4*)(p + 16);         // global_load_b128
  return f.v;
}

__device__ __forceinline__ void store_tile(v8f c, float* C, int rb, int cb,
                                           int M, int N, const float* bias,
                                           int relu, int lane) {
  int col = cb + (lane & 15);
  if (col >= N) return;
  float bv = bias ? bias[col] : 0.f;
  int r0 = rb + (lane >> 4) * 8;
  #pragma unroll
  for (int r = 0; r < 8; r++) {
    int row = r0 + r;
    if (row < M) {
      float v = c[r] + bv;
      if (relu) v = fmaxf(v, 0.f);
      C[(long)row * N + col] = v;
    }
  }
}

// ---- elementwise kernels ---------------------------------------------------

__global__ void k_cast_bf16(const float* __restrict__ in,
                            __bf16* __restrict__ out, long n) {
  long i = (long)blockIdx.x * blockDim.x + threadIdx.x;
  long s = (long)gridDim.x * blockDim.x;
  for (; i < n; i += s) out[i] = f2bf(in[i]);
}

__global__ void k_zero(float* __restrict__ p, long n) {
  long i = (long)blockIdx.x * blockDim.x + threadIdx.x;
  long s = (long)gridDim.x * blockDim.x;
  for (; i < n; i += s) p[i] = 0.f;
}

__global__ void k_sigmoid(float* __restrict__ x, long n, float scale) {
  long i = (long)blockIdx.x * blockDim.x + threadIdx.x;
  long s = (long)gridDim.x * blockDim.x;
  for (; i < n; i += s) x[i] = 1.f / (1.f + __expf(-x[i] * scale));
}

// out[i] = bf16(w[i/D] * in[i])  (ta-scaled text -> bf16)
__global__ void k_scale_cast(const float* __restrict__ in,
                             const float* __restrict__ w,
                             __bf16* __restrict__ out, long n, int d) {
  long i = (long)blockIdx.x * blockDim.x + threadIdx.x;
  long s = (long)gridDim.x * blockDim.x;
  for (; i < n; i += s) out[i] = f2bf(w[i / d] * in[i]);
}

// CLS[b,d] = (1/N) * sum_n w[b,n] * X[b,n,d]
__global__ void k_scale_reduce(const float* __restrict__ X,
                               const float* __restrict__ w,
                               float* __restrict__ out, int N) {
  int d = blockIdx.x * 256 + threadIdx.x;   // 0..767
  int b = blockIdx.y;
  float acc = 0.f;
  const float* xb = X + (long)b * N * DD + d;
  const float* wb = w + (long)b * N;
  for (int n = 0; n < N; n++) acc += wb[n] * xb[(long)n * DD];
  out[b * DD + d] = acc / (float)N;
}

// ---- generic batched bf16 WMMA GEMM:  C = A(MxK) * B(NxK)^T (+bias)(+relu)
// 256 threads = 8 waves arranged 4(M) x 2(N); wave tile 32x64 (2x4 WMMA);
// WG tile 128x128. Per k-step: 6 fragment loads feed 8 WMMAs.
__global__ __launch_bounds__(256)
void k_gemm_bf16(const __bf16* __restrict__ A, const __bf16* __restrict__ Bm,
                 float* __restrict__ C, int M, int N, int K,
                 long sA, long sB, long sC, const float* __restrict__ bias,
                 int relu) {
  int lane = threadIdx.x & 31;
  int w    = threadIdx.x >> 5;
  int b    = blockIdx.z;
  const __bf16* Ab = A + (long)b * sA;
  const __bf16* Bb = Bm + (long)b * sB;
  float* Cb = C + (long)b * sC;
  int rowBase = blockIdx.x * 128 + (w >> 1) * 32;
  int colBase = blockIdx.y * 128 + (w & 1) * 64;

  v8f acc[2][4];
  #pragma unroll
  for (int m = 0; m < 2; m++)
    #pragma unroll
    for (int n = 0; n < 4; n++) acc[m][n] = (v8f){};

  for (int kb = 0; kb < K; kb += 32) {
    v16bf a0 = load_frag(Ab, rowBase,      M, K, kb, lane);
    v16bf a1 = load_frag(Ab, rowBase + 16, M, K, kb, lane);
    v16bf b0 = load_frag(Bb, colBase,      N, K, kb, lane);
    v16bf b1 = load_frag(Bb, colBase + 16, N, K, kb, lane);
    v16bf b2 = load_frag(Bb, colBase + 32, N, K, kb, lane);
    v16bf b3 = load_frag(Bb, colBase + 48, N, K, kb, lane);
    acc[0][0] = wmma_bf16(a0, b0, acc[0][0]);
    acc[0][1] = wmma_bf16(a0, b1, acc[0][1]);
    acc[0][2] = wmma_bf16(a0, b2, acc[0][2]);
    acc[0][3] = wmma_bf16(a0, b3, acc[0][3]);
    acc[1][0] = wmma_bf16(a1, b0, acc[1][0]);
    acc[1][1] = wmma_bf16(a1, b1, acc[1][1]);
    acc[1][2] = wmma_bf16(a1, b2, acc[1][2]);
    acc[1][3] = wmma_bf16(a1, b3, acc[1][3]);
  }
  #pragma unroll
  for (int m = 0; m < 2; m++)
    #pragma unroll
    for (int n = 0; n < 4; n++)
      store_tile(acc[m][n], Cb, rowBase + 16 * m, colBase + 16 * n,
                 M, N, bias, relu, lane);
}

// ---- fused conv1(1->64,3x3)+conv2(64->1,3x3)+ReLU + row/col sum reduction
// one 16x16 output tile per 256-thread WG, exact SAME zero-padding at both
// conv stages. Accumulates sum_t -> va_logit[b,v], sum_v -> ta_logit[b,t].
__global__ __launch_bounds__(256)
void k_conv(const float* __restrict__ S, const float* __restrict__ w1,
            const float* __restrict__ b1, const float* __restrict__ w2,
            const float* __restrict__ b2, float* __restrict__ va_logit,
            float* __restrict__ ta_logit) {
  __shared__ float xs[20][20];
  __shared__ float ys[18][20];
  __shared__ float zs[16][16];
  __shared__ float w1s[64 * 9];
  __shared__ float w2s[64 * 9];
  __shared__ float b1s[64];

  int tx = threadIdx.x, ty = threadIdx.y;
  int tid = ty * 16 + tx;
  int b = blockIdx.z;
  int v0 = blockIdx.x * 16;
  int t0 = blockIdx.y * 16;

  for (int i = tid; i < 576; i += 256) { w1s[i] = w1[i]; w2s[i] = w2[i]; }
  if (tid < 64) b1s[tid] = b1[tid];

  const float* Sb = S + (long)b * VV * TT;
  for (int i = tid; i < 400; i += 256) {
    int lv = i / 20, lt = i % 20;
    int v = v0 + lv - 2, t = t0 + lt - 2;
    xs[lv][lt] = (v >= 0 && v < VV && t >= 0 && t < TT)
                     ? Sb[(long)v * TT + t] : 0.f;
  }
  __syncthreads();

  float z = 0.f;
  for (int c = 0; c < 64; c++) {
    // conv1 plane (tile + 1 halo), zero outside the image (conv2 SAME pad)
    for (int i = tid; i < 324; i += 256) {
      int ly = i / 18, lt = i % 18;
      int vy = v0 + ly - 1, typ = t0 + lt - 1;
      float y = 0.f;
      if (vy >= 0 && vy < VV && typ >= 0 && typ < TT) {
        y = b1s[c];
        #pragma unroll
        for (int ii = 0; ii < 3; ii++)
          #pragma unroll
          for (int jj = 0; jj < 3; jj++)
            y += w1s[c * 9 + ii * 3 + jj] * xs[ly + ii][lt + jj];
      }
      ys[ly][lt] = y;
    }
    __syncthreads();
    float a = 0.f;
    #pragma unroll
    for (int ii = 0; ii < 3; ii++)
      #pragma unroll
      for (int jj = 0; jj < 3; jj++)
        a += w2s[c * 9 + ii * 3 + jj] * ys[ty + ii][tx + jj];
    z += a;
    __syncthreads();
  }
  z = fmaxf(z + b2[0], 0.f);

  int vo = v0 + ty, to = t0 + tx;
  if (vo >= VV) z = 0.f;
  zs[ty][tx] = z;
  __syncthreads();
  if (tx == 0 && vo < VV) {
    float rs = 0.f;
    #pragma unroll
    for (int j = 0; j < 16; j++) rs += zs[ty][j];
    atomicAdd(&va_logit[b * VV + vo], rs);
  }
  if (ty == 0) {
    float cs = 0.f;
    #pragma unroll
    for (int i = 0; i < 16; i++) cs += zs[i][tx];
    atomicAdd(&ta_logit[b * TT + to], cs);
  }
}

// ---- pred + masked squared error accumulation (one wave per (b,t)) --------
__global__ void k_pred(const float* __restrict__ H,
                       const float* __restrict__ fc2_w,
                       const float* __restrict__ fc2_b,
                       const float* __restrict__ sent,
                       const int* __restrict__ lens,
                       float* __restrict__ acc) {
  int t = blockIdx.x, b = blockIdx.y, lane = threadIdx.x;
  const float* h = H + ((long)b * TT + t) * DD;
  float d = 0.f;
  for (int n = lane; n < DD; n += 32) d += fc2_w[n] * h[n];
  d = wred32(d);
  if (lane == 0) {
    float p = d + fc2_b[0];
    float s = sent[b * TT + t];
    p = (s != 0.f) ? p : 0.f;                 // zmask
    if (t < lens[b]) {
      float e = p - s;
      atomicAdd(&acc[b], e * e);
    }
  }
}

// ---- the B=32 scalar soup: one 256-thread workgroup -----------------------
__global__ __launch_bounds__(256)
void k_final(const float* __restrict__ v_emb, const float* __restrict__ t_emb,
             const float* __restrict__ vCLS, const float* __restrict__ tCLS,
             const float* __restrict__ fc2_w, const float* __restrict__ fc2_b,
             const float* __restrict__ final_w, const float* __restrict__ final_b,
             const float* __restrict__ pred_acc, const int* __restrict__ lens,
             float* __restrict__ out) {
  __shared__ float vs[32], ts_[32], nv[32], nt[32];
  __shared__ float sim[32][32];
  __shared__ float srow[32], crow[32];
  __shared__ float varv[DD], vart[DD];
  __shared__ float red[256];
  __shared__ float s_invnv, s_invnt;

  int tid = threadIdx.x, lane = tid & 31, wave = tid >> 5;

  // A: v_s, t_s, row norms (4 rows per wave)
  for (int r = 0; r < 4; r++) {
    int i = wave * 4 + r;
    float sv = 0.f, st = 0.f, n2v = 0.f, n2t = 0.f;
    for (int n = lane; n < DD; n += 32) {
      float ve = v_emb[i * DD + n], te = t_emb[i * DD + n];
      sv += fc2_w[n] * fmaxf(ve, 0.f);
      st += fc2_w[n] * te;
      n2v += ve * ve;
      n2t += te * te;
    }
    sv = wred32(sv); st = wred32(st); n2v = wred32(n2v); n2t = wred32(n2t);
    if (lane == 0) {
      vs[i] = sv + fc2_b[0];
      ts_[i] = st + fc2_b[0];
      nv[i] = fmaxf(sqrtf(n2v), 1e-12f);
      nt[i] = fmaxf(sqrtf(n2t), 1e-12f);
    }
  }
  __syncthreads();

  // C: sim[i][j] = exp( (v_emb_i . t_emb_j) / (|v_i||t_j| * 0.2) )
  for (int p = 0; p < 128; p++) {
    int pp = wave * 128 + p;
    int i = pp >> 5, j = pp & 31;
    float d = 0.f;
    for (int n = lane; n < DD; n += 32)
      d += v_emb[i * DD + n] * t_emb[j * DD + n];
    d = wred32(d);
    if (lane == 0) sim[i][j] = __expf(d / (nv[i] * nt[j] * 0.2f));
  }
  __syncthreads();

  // D: row sums + KL-style contrast loss
  if (tid < 32) {
    float sr = 0.f, cr = 0.f;
    for (int j = 0; j < 32; j++) {
      sr += sim[tid][j];
      cr += __expf(-fabsf(vs[tid] - ts_[j]));
    }
    srow[tid] = sr;
    crow[tid] = cr;
  }
  __syncthreads();
  {
    float lacc = 0.f;
    for (int p = tid; p < 1024; p += 256) {
      int i = p >> 5, j = p & 31;
      float c = __expf(-fabsf(vs[i] - ts_[j])) / crow[i];
      float s = sim[i][j] / srow[i];
      lacc += c * (__logf(c) - __logf(s));
    }
    red[tid] = lacc;
    __syncthreads();
    for (int s = 128; s > 0; s >>= 1) {
      if (tid < s) red[tid] += red[tid + s];
      __syncthreads();
    }
    if (tid == 0) out[32] = red[0] / 32.f;
    __syncthreads();
  }

  // E: per-dim ddof=1 variance over batch, then L2-normalize the var vectors
  float pv = 0.f, pt = 0.f;
  for (int d = tid; d < DD; d += 256) {
    float mv = 0.f, mt = 0.f;
    for (int bb = 0; bb < 32; bb++) {
      mv += vCLS[bb * DD + d];
      mt += tCLS[bb * DD + d];
    }
    mv *= (1.f / 32.f);
    mt *= (1.f / 32.f);
    float vv = 0.f, vt = 0.f;
    for (int bb = 0; bb < 32; bb++) {
      float e = vCLS[bb * DD + d] - mv; vv += e * e;
      e = tCLS[bb * DD + d] - mt;       vt += e * e;
    }
    vv *= (1.f / 31.f);
    vt *= (1.f / 31.f);
    varv[d] = vv; vart[d] = vt;
    pv += vv * vv; pt += vt * vt;
  }
  red[tid] = pv;
  __syncthreads();
  for (int s = 128; s > 0; s >>= 1) {
    if (tid < s) red[tid] += red[tid + s];
    __syncthreads();
  }
  if (tid == 0) s_invnv = 1.f / fmaxf(sqrtf(red[0]), 1e-12f);
  __syncthreads();
  red[tid] = pt;
  __syncthreads();
  for (int s = 128; s > 0; s >>= 1) {
    if (tid < s) red[tid] += red[tid + s];
    __syncthreads();
  }
  if (tid == 0) s_invnt = 1.f / fmaxf(sqrtf(red[0]), 1e-12f);
  __syncthreads();

  // F/G: fusion head + final projection + lambda offset
  for (int r = 0; r < 4; r++) {
    int b = wave * 4 + r;
    float acc = 0.f;
    for (int d = lane; d < DD; d += 32) {
      float sv_ = vCLS[b * DD + d] * (1.f + varv[d] * s_invnv);
      float st_ = tCLS[b * DD + d] * (1.f + vart[d] * s_invnt);
      float fcls = (sv_ * st_ + v_emb[b * DD + d] * t_emb[b * DD + d]) * 0.5f;
      acc += final_w[d] * fcls;
    }
    acc = wred32(acc);
    if (lane == 0)
      out[b] = acc + final_b[0] + 0.1f * fabsf(ts_[b] - vs[b]) - 0.5f;
  }

  // text sentiment loss
  if (tid == 0) {
    float a = 0.f;
    for (int bb = 0; bb < 32; bb++) a += pred_acc[bb] / (float)lens[bb];
    out[33] = a / 32.f;
  }
}

// ---------------------------------------------------------------------------

extern "C" void kernel_launch(void* const* d_in, const int* in_sizes, int n_in,
                              void* d_out, int out_size, void* d_ws, size_t ws_size,
                              hipStream_t stream) {
  const float* vision = (const float*)d_in[0];
  const float* text   = (const float*)d_in[1];
  const float* sent   = (const float*)d_in[2];
  const int*   lens   = (const int*)d_in[3];
  const float* fc1_w  = (const float*)d_in[4];
  const float* fc1_b  = (const float*)d_in[5];
  const float* fc2_w  = (const float*)d_in[6];
  const float* fc2_b  = (const float*)d_in[7];
  const float* c1w    = (const float*)d_in[8];
  const float* c1b    = (const float*)d_in[9];
  const float* c2w    = (const float*)d_in[10];
  const float* c2b    = (const float*)d_in[11];
  const float* fw     = (const float*)d_in[12];
  const float* fb     = (const float*)d_in[13];
  float* out = (float*)d_out;

  char* ws = (char*)d_ws;
  size_t off = 0;
  auto alloc = [&](size_t bytes) -> void* {
    void* p = ws + off;
    off += (bytes + 255) & ~(size_t)255;
    return p;
  };

  __bf16* visB  = (__bf16*)alloc((size_t)BB * VV * DD * 2);
  __bf16* textB = (__bf16*)alloc((size_t)BB * TT * DD * 2);  // later: ta-scaled txt
  __bf16* fc1B  = (__bf16*)alloc((size_t)DD * DD * 2);
  float*  S     = (float*) alloc((size_t)BB * VV * TT * 4);
  float*  va    = (float*) alloc((size_t)BB * VV * 4);
  float*  ta    = (float*) alloc((size_t)BB * TT * 4);
  float*  pacc  = (float*) alloc((size_t)BB * 4);
  float*  vCLS  = (float*) alloc((size_t)BB * DD * 4);
  float*  tCLS  = (float*) alloc((size_t)BB * DD * 4);
  __bf16* vCLSb = (__bf16*)alloc((size_t)BB * DD * 2);
  __bf16* tCLSb = (__bf16*)alloc((size_t)BB * DD * 2);
  float*  H     = (float*) alloc((size_t)BB * TT * DD * 4);
  float*  v_emb = (float*) alloc((size_t)BB * DD * 4);
  float*  t_emb = (float*) alloc((size_t)BB * DD * 4);

  // 1) fp32 -> bf16 operand casts
  k_cast_bf16<<<4096, 256, 0, stream>>>(vision, visB, (long)BB * VV * DD);
  k_cast_bf16<<<8192, 256, 0, stream>>>(text, textB, (long)BB * TT * DD);
  k_cast_bf16<<<2304, 256, 0, stream>>>(fc1_w, fc1B, (long)DD * DD);

  // 2) batched WMMA GEMM: S[b] = vision[b] (197x768) * text[b]^T (768x512)
  k_gemm_bf16<<<dim3(2, 4, BB), 256, 0, stream>>>(
      visB, textB, S, VV, TT, DD,
      (long)VV * DD, (long)TT * DD, (long)VV * TT, nullptr, 0);

  // 3) zero the atomic accumulators (ws is poisoned)
  k_zero<<<32, 256, 0, stream>>>(va, (long)BB * VV);
  k_zero<<<64, 256, 0, stream>>>(ta, (long)BB * TT);
  k_zero<<<1, 32, 0, stream>>>(pacc, BB);

  // 4) fused conv1+conv2+ReLU + row/col sums
  k_conv<<<dim3(13, 32, BB), dim3(16, 16), 0, stream>>>(S, c1w, c1b, c2w, c2b,
                                                        va, ta);

  // 5) sigmoid(sum/T), sigmoid(sum/V)
  k_sigmoid<<<25, 256, 0, stream>>>(va, (long)BB * VV, 1.f / (float)TT);
  k_sigmoid<<<64, 256, 0, stream>>>(ta, (long)BB * TT, 1.f / (float)VV);

  // 6) CLS means of the attention-scaled embeddings
  k_scale_reduce<<<dim3(3, BB), 256, 0, stream>>>(vision, va, vCLS, VV);
  k_scale_reduce<<<dim3(3, BB), 256, 0, stream>>>(text, ta, tCLS, TT);

  // 7) txt = ta * text in bf16 (overwrites textB; GEMM #2 is already done)
  k_scale_cast<<<8192, 256, 0, stream>>>(text, ta, textB, (long)BB * TT * DD, DD);

  // 8) CLS -> bf16
  k_cast_bf16<<<96, 256, 0, stream>>>(vCLS, vCLSb, (long)BB * DD);
  k_cast_bf16<<<96, 256, 0, stream>>>(tCLS, tCLSb, (long)BB * DD);

  // 9) H = relu(txt @ fc1^T + fc1_b)  (16384 x 768 x 768, the big GEMM)
  k_gemm_bf16<<<dim3(128, 6, 1), 256, 0, stream>>>(
      textB, fc1B, H, BB * TT, DD, DD, 0, 0, 0, fc1_b, 1);

  // 10) t_emb / v_emb = CLS @ fc1^T + fc1_b (no relu)
  k_gemm_bf16<<<dim3(1, 6, 1), 256, 0, stream>>>(
      tCLSb, fc1B, t_emb, BB, DD, DD, 0, 0, 0, fc1_b, 0);
  k_gemm_bf16<<<dim3(1, 6, 1), 256, 0, stream>>>(
      vCLSb, fc1B, v_emb, BB, DD, DD, 0, 0, 0, fc1_b, 0);

  // 11) pred + masked MSE accumulation
  k_pred<<<dim3(TT, BB), 32, 0, stream>>>(H, fc2_w, fc2_b, sent, lens, pacc);

  // 12) contrast loss, variance branch, fusion head, outputs
  k_final<<<1, 256, 0, stream>>>(v_emb, t_emb, vCLS, tCLS, fc2_w, fc2_b, fw, fb,
                                 pacc, lens, out);
}
